// BahdanauAttention_48945447306027
// MI455X (gfx1250) — matrix operand order repaired and checked by reading, not compiled, not run on hardware
//
#include <hip/hip_runtime.h>

typedef __attribute__((ext_vector_type(16))) __bf16 v16bf;
typedef __attribute__((ext_vector_type(8)))  float  v8f;

#define DIM 1024
#define SEQ 2048
#define NBATCH 32
#define MT 64              // rows (s positions) per workgroup
#define LDS_STRIDE 1032    // bf16 elems per row, +16B pad -> conflict-free ds_load_b128

__device__ __forceinline__ unsigned short f32_bf16_bits(float f) {
  union { float f; unsigned u; } x; x.f = f;
  unsigned r = x.u + 0x7FFFu + ((x.u >> 16) & 1u);   // round-to-nearest-even
  return (unsigned short)(r >> 16);
}

// Branch-free tanh: prefer the CDNA5 V_TANH trans op (co-executes with WMMA);
// fallback is exec-mask-free exp/rcp formulation with correct +/-1 saturation.
__device__ __forceinline__ float fast_tanhf(float x) {
#if __has_builtin(__builtin_amdgcn_tanhf)
  return __builtin_amdgcn_tanhf(x);
#elif __has_builtin(__builtin_amdgcn_tanh_f32)
  return __builtin_amdgcn_tanh_f32(x);
#else
  const float e = __expf(2.0f * x);                 // v_exp_f32
  return 1.0f - 2.0f * __builtin_amdgcn_rcpf(e + 1.0f);  // v_rcp_f32
#endif
}

// ---------------------------------------------------------------------------
// Kernel 1: W [d][e] f32  ->  Wt [e][d] bf16 (transposed, for B-fragment loads)
// ---------------------------------------------------------------------------
__global__ void __launch_bounds__(256)
wconv_kernel(const float* __restrict__ W, unsigned short* __restrict__ Wt) {
  __shared__ unsigned short tile[32][33];
  const int bx = blockIdx.x * 32;   // e tile
  const int by = blockIdx.y * 32;   // d tile
  const int tx = threadIdx.x, ty = threadIdx.y;   // 32 x 8
#pragma unroll
  for (int j = 0; j < 32; j += 8)
    tile[ty + j][tx] = f32_bf16_bits(W[(size_t)(by + ty + j) * DIM + bx + tx]);
  __syncthreads();
#pragma unroll
  for (int j = 0; j < 32; j += 8)
    Wt[(size_t)(bx + ty + j) * DIM + by + tx] = tile[tx][ty + j];
}

// ---------------------------------------------------------------------------
// Kernel 2: fused score = U . tanh(enc @ W + B), bf16 WMMA GEMM
// grid: 1024 workgroups (32 batches x 32 s-tiles of 64), block: 256 (8 waves)
// ---------------------------------------------------------------------------
__global__ void __launch_bounds__(256)
score_gemm_kernel(const float* __restrict__ enc,
                  const unsigned short* __restrict__ Wt,
                  const float* __restrict__ Bv,
                  const float* __restrict__ Uv,
                  float* __restrict__ score) {
  __shared__ unsigned short A_lds[MT * LDS_STRIDE];  // 64 x 1024 bf16 (padded)
  __shared__ float score_lds[MT];

  const int tid   = threadIdx.x;
  const int lane  = tid & 31;
  const int wave  = tid >> 5;
  const int blk   = blockIdx.x;
  const int b     = blk >> 5;            // batch
  const int sBase = (blk & 31) * MT;     // seq start
  const size_t rowBase = ((size_t)b * SEQ + sBase) * DIM;

  if (tid < MT) score_lds[tid] = 0.0f;

  // Stage encoded block [MT x DIM] into LDS as bf16 (single HBM pass for GEMM)
  for (int i = tid; i < (MT * DIM) / 4; i += 256) {
    const int e4  = i << 2;
    const int row = e4 >> 10;
    const int col = e4 & (DIM - 1);
    const float4 v = *(const float4*)(enc + rowBase + ((size_t)row << 10) + col);
    uint2 p;
    p.x = (unsigned)f32_bf16_bits(v.x) | ((unsigned)f32_bf16_bits(v.y) << 16);
    p.y = (unsigned)f32_bf16_bits(v.z) | ((unsigned)f32_bf16_bits(v.w) << 16);
    *(uint2*)&A_lds[row * LDS_STRIDE + col] = p;
  }
  __syncthreads();

  const int wm    = wave >> 1;        // 0..3 : M row-block
  const int wn    = wave & 1;         // 0..1 : N half within 32-wide N tile
  const int m0    = wm * 16;
  const int laneL = lane & 15;        // A: row within tile / B,C: column within tile
  const int laneH = lane >> 4;        // lane half
  const int khalf = laneH * 8;        // A-fragment K sub-block per ISA layout

  float rowacc[8];
#pragma unroll
  for (int j = 0; j < 8; ++j) rowacc[j] = 0.0f;

  const unsigned short* Arow = &A_lds[(m0 + laneL) * LDS_STRIDE];

  for (int nIt = 0; nIt < DIM / 32; ++nIt) {
    const int n0 = nIt * 32 + wn * 16;
    const unsigned short* Brow = Wt + (size_t)(n0 + laneL) * DIM;
    const float bias = Bv[n0 + laneL];
    const float uval = Uv[n0 + laneL];

    // prefetch next B panel (speculative; stays within d_ws)
    __builtin_prefetch(Wt + (size_t)(n0 + 32 + laneL) * DIM, 0, 1);

    // Two accumulators: halves the WMMA->WMMA dependency chain in the K loop
    v8f acc0 = {};
    v8f acc1 = {};
#pragma unroll 4
    for (int k = 0; k < DIM; k += 32) {
      union { uint4 q[2]; v16bf v; } af, bfr;
      // A 16x32 bf16: lanes 0-15 hold K{0..7,16..23}, lanes 16-31 K{8..15,24..31}
      af.q[0] = *(const uint4*)(Arow + k + khalf);
      af.q[1] = *(const uint4*)(Arow + k + 16 + khalf);
      // B 32x16 bf16: column = laneL; lanes 0-15 K{0..7,16..23}, lanes 16-31 K{8..15,24..31}
      const int kb = k + laneH * 8;
      bfr.q[0] = *(const uint4*)(Brow + kb);
      bfr.q[1] = *(const uint4*)(Brow + kb + 16);
      if ((k >> 5) & 1) {
        acc1 = __builtin_amdgcn_wmma_f32_16x16x32_bf16(
            false, af.v, false, bfr.v, (short)0, acc1, false, false);
      } else {
        acc0 = __builtin_amdgcn_wmma_f32_16x16x32_bf16(
            false, af.v, false, bfr.v, (short)0, acc0, false, false);
      }
    }

    // Fused epilogue: lane holds column n0+laneL, rows m0 + laneH*8 + j
#pragma unroll
    for (int j = 0; j < 8; ++j) {
      const float t = fast_tanhf(acc0[j] + acc1[j] + bias);
      rowacc[j] += t * uval;
    }
  }

  // Reduce each row across the 16 lanes of its half, then across the 2 wn waves
#pragma unroll
  for (int j = 0; j < 8; ++j) {
    float v = rowacc[j];
    v += __shfl_xor(v, 1);
    v += __shfl_xor(v, 2);
    v += __shfl_xor(v, 4);
    v += __shfl_xor(v, 8);
    if (laneL == 0) atomicAdd(&score_lds[m0 + laneH * 8 + j], v);  // ds_add_f32
  }
  __syncthreads();
  if (tid < MT) score[(size_t)b * SEQ + sBase + tid] = score_lds[tid];
}

// ---------------------------------------------------------------------------
// Kernel 3: per-batch softmax over SEQ=2048 (one block per batch)
// ---------------------------------------------------------------------------
__global__ void __launch_bounds__(256)
softmax_kernel(const float* __restrict__ score, float* __restrict__ alpha) {
  const int b = blockIdx.x;
  const int tid = threadIdx.x;
  const int lane = tid & 31, wave = tid >> 5;
  __shared__ float red[8];

  float v[8];
  float m = -3.4e38f;
#pragma unroll
  for (int i = 0; i < 8; ++i) {
    v[i] = score[(size_t)b * SEQ + tid + i * 256];
    m = fmaxf(m, v[i]);
  }
#pragma unroll
  for (int o = 16; o >= 1; o >>= 1) m = fmaxf(m, __shfl_xor(m, o));
  if (lane == 0) red[wave] = m;
  __syncthreads();
  float gm = red[0];
#pragma unroll
  for (int w = 1; w < 8; ++w) gm = fmaxf(gm, red[w]);
  __syncthreads();

  float s = 0.0f;
#pragma unroll
  for (int i = 0; i < 8; ++i) { v[i] = __expf(v[i] - gm); s += v[i]; }
#pragma unroll
  for (int o = 16; o >= 1; o >>= 1) s += __shfl_xor(s, o);
  if (lane == 0) red[wave] = s;
  __syncthreads();
  float gs = 0.0f;
#pragma unroll
  for (int w = 0; w < 8; ++w) gs += red[w];
  const float inv = 1.0f / gs;
#pragma unroll
  for (int i = 0; i < 8; ++i)
    alpha[(size_t)b * SEQ + tid + i * 256] = v[i] * inv;
}

// ---------------------------------------------------------------------------
// Kernel 4: context[b][d] = sum_s alpha[b][s] * enc[b][s][d]  (bandwidth-bound)
// grid: (DIM/256, NBATCH), block 256 -> coalesced 1KB rows per s
// ---------------------------------------------------------------------------
__global__ void __launch_bounds__(256)
context_kernel(const float* __restrict__ enc,
               const float* __restrict__ alpha,
               float* __restrict__ out) {
  const int d = blockIdx.x * 256 + threadIdx.x;
  const int b = blockIdx.y;
  const float* ep = enc + (size_t)b * SEQ * DIM + d;
  const float* ap = alpha + (size_t)b * SEQ;
  float acc = 0.0f;
  for (int s = 0; s < SEQ; s += 4) {
    const float a0 = ap[s + 0], a1 = ap[s + 1], a2 = ap[s + 2], a3 = ap[s + 3];
    acc += ep[(size_t)(s + 0) * DIM] * a0;
    acc += ep[(size_t)(s + 1) * DIM] * a1;
    acc += ep[(size_t)(s + 2) * DIM] * a2;
    acc += ep[(size_t)(s + 3) * DIM] * a3;
  }
  out[(size_t)b * DIM + d] = acc;
}

// ---------------------------------------------------------------------------
extern "C" void kernel_launch(void* const* d_in, const int* in_sizes, int n_in,
                              void* d_out, int out_size, void* d_ws, size_t ws_size,
                              hipStream_t stream) {
  const float* enc = (const float*)d_in[0];   // [32,2048,1024]
  const float* W   = (const float*)d_in[1];   // [1024,1024]
  const float* Bv  = (const float*)d_in[2];   // [1024]
  const float* Uv  = (const float*)d_in[3];   // [1024]
  float* out = (float*)d_out;                 // [32,1024]

  // workspace layout: Wt bf16 (2MB) | score (256KB) | alpha (256KB)
  unsigned short* Wt = (unsigned short*)d_ws;
  float* score = (float*)((char*)d_ws + (size_t)DIM * DIM * sizeof(unsigned short));
  float* alpha = score + (size_t)NBATCH * SEQ;

  wconv_kernel<<<dim3(DIM / 32, DIM / 32), dim3(32, 8), 0, stream>>>(W, Wt);
  score_gemm_kernel<<<dim3(NBATCH * (SEQ / MT)), 256, 0, stream>>>(enc, Wt, Bv, Uv, score);
  softmax_kernel<<<NBATCH, 256, 0, stream>>>(score, alpha);
  context_kernel<<<dim3(DIM / 256, NBATCH), 256, 0, stream>>>(enc, alpha, out);
}
